// Transformer_16441134809861
// MI455X (gfx1250) — matrix operand order repaired and checked by reading, not compiled, not run on hardware
//
#include <hip/hip_runtime.h>
#include <math.h>

// ---------------- problem constants ----------------
#define DEPTH 2
#define DIMX 1024
#define HEADS 16
#define DHD 64
#define BB 8
#define NN 256
#define JJ 1088      // MEM*N + TEXT = 1024 + 64
#define MS 1024      // memory token count
#define FFH 4096     // DIM*4
#define SCALE 0.125f // DH^-0.5

typedef __attribute__((ext_vector_type(16))) __bf16 bfvec16;
typedef __attribute__((ext_vector_type(8)))  float  fvec8;

union FragB { bfvec16 v; uint4 q[2]; };

__device__ inline fvec8 fzero() {
  fvec8 z = {0.f,0.f,0.f,0.f,0.f,0.f,0.f,0.f};
  return z;
}

__device__ inline fvec8 wmma_bf16(const FragB& a, const FragB& b, fvec8 c) {
  return __builtin_amdgcn_wmma_f32_16x16x32_bf16(
      false, a.v, false, b.v, (short)0, c, false, false);
}

__device__ inline unsigned short f2bf(float f) {
  unsigned int u = __float_as_uint(f);
  u += 0x7FFFu + ((u >> 16) & 1u);   // round-to-nearest-even
  return (unsigned short)(u >> 16);
}

// ---------------- CDNA5 async global->LDS path ----------------
#if defined(__has_builtin)
#if __has_builtin(__builtin_amdgcn_global_load_async_to_lds_b128)
#define HAVE_ASYNC_LDS 1
#endif
#if __has_builtin(__builtin_amdgcn_s_wait_asynccnt)
#define HAVE_WAIT_ASYNC 1
#endif
#endif

typedef int vi4_t __attribute__((vector_size(16)));
typedef __attribute__((address_space(1))) vi4_t* gvi4_p;
typedef __attribute__((address_space(3))) vi4_t* lvi4_p;

__device__ inline void async_load_b128(const void* g, void* l) {
#if defined(HAVE_ASYNC_LDS)
  __builtin_amdgcn_global_load_async_to_lds_b128((gvi4_p)g, (lvi4_p)l, 0, 0);
#else
  *(uint4*)l = *(const uint4*)g;   // sync fallback through VGPRs
#endif
}

template <int N>
__device__ inline void wait_asynccnt() {
#if defined(HAVE_ASYNC_LDS)
#if defined(HAVE_WAIT_ASYNC)
  __builtin_amdgcn_s_wait_asynccnt((short)N);
#else
  asm volatile("s_wait_asynccnt %0" ::"n"(N) : "memory");
#endif
#endif
}

// ---------------- elementwise kernels ----------------
__global__ void f2bf_copy_kernel(const float* __restrict__ in,
                                 unsigned short* __restrict__ out, size_t n) {
  size_t i = (size_t)blockIdx.x * blockDim.x + threadIdx.x;
  if (i < n) out[i] = f2bf(in[i]);
}

// in [K, Nc] fp32 -> out [Nc, K] bf16
__global__ void f2bf_transpose_kernel(const float* __restrict__ in,
                                      unsigned short* __restrict__ out,
                                      int K, int Nc) {
  size_t n = (size_t)K * Nc;
  size_t i = (size_t)blockIdx.x * blockDim.x + threadIdx.x;
  if (i >= n) return;
  int k = (int)(i / Nc);
  int c = (int)(i % Nc);
  out[(size_t)c * K + k] = f2bf(in[i]);
}

// rotary on q: q fp32 [B*N, 1024] -> Qo bf16 [B,H,N,DH]
__global__ void rope_q_kernel(const float* __restrict__ q,
                              const float* __restrict__ src_freq,
                              unsigned short* __restrict__ Qo) {
  size_t total = (size_t)BB * NN * HEADS * (DHD / 2);
  size_t i = (size_t)blockIdx.x * blockDim.x + threadIdx.x;
  if (i >= total) return;
  int pr = (int)(i % (DHD / 2));
  int h  = (int)((i / (DHD / 2)) % HEADS);
  int n  = (int)((i / ((DHD / 2) * HEADS)) % NN);
  int b  = (int)(i / ((size_t)(DHD / 2) * HEADS * NN));
  int d0 = pr * 2;
  const float* row = q + ((size_t)b * NN + n) * DIMX + h * DHD;
  float t1 = row[d0], t2 = row[d0 + 1];
  float f0 = src_freq[n * DHD + d0], f1 = src_freq[n * DHD + d0 + 1];
  float o0 = t1 * cosf(f0) - t2 * sinf(f0);
  float o1 = t2 * cosf(f1) + t1 * sinf(f1);
  size_t ob = (((size_t)(b * HEADS + h)) * NN + n) * DHD;
  Qo[ob + d0]     = f2bf(o0);
  Qo[ob + d0 + 1] = f2bf(o1);
}

// bias + rotary on k: k fp32 [B*J, 1024] -> Ko bf16 [B,H,J,DH]
__global__ void biasrope_k_kernel(const float* __restrict__ kin,
                                  const float* __restrict__ tgt_freq,
                                  const float* __restrict__ rpe_l,
                                  const int* __restrict__ rel_idx,
                                  unsigned short* __restrict__ Ko) {
  size_t total = (size_t)BB * JJ * HEADS * (DHD / 2);
  size_t i = (size_t)blockIdx.x * blockDim.x + threadIdx.x;
  if (i >= total) return;
  int pr = (int)(i % (DHD / 2));
  int h  = (int)((i / (DHD / 2)) % HEADS);
  int j  = (int)((i / ((DHD / 2) * HEADS)) % JJ);
  int b  = (int)(i / ((size_t)(DHD / 2) * HEADS * JJ));
  int d0 = pr * 2;
  const float* row = kin + ((size_t)b * JJ + j) * DIMX + h * DHD;
  float t1 = row[d0], t2 = row[d0 + 1];
  float o0, o1;
  if (j < MS) {
    float bias = rpe_l[rel_idx[j >> 8] * HEADS + h];
    t1 += bias; t2 += bias;
    float f0 = tgt_freq[j * DHD + d0], f1 = tgt_freq[j * DHD + d0 + 1];
    o0 = t1 * cosf(f0) - t2 * sinf(f0);
    o1 = t2 * cosf(f1) + t1 * sinf(f1);
  } else {
    o0 = t1; o1 = t2;
  }
  size_t ob = (((size_t)(b * HEADS + h)) * JJ + j) * DHD;
  Ko[ob + d0]     = f2bf(o0);
  Ko[ob + d0 + 1] = f2bf(o1);
}

// v bf16 [B*J, 1024] -> Vt bf16 [B,H,DH,J]
__global__ void vtrans_kernel(const unsigned short* __restrict__ vin,
                              unsigned short* __restrict__ Vt) {
  size_t total = (size_t)BB * JJ * DIMX;
  size_t i = (size_t)blockIdx.x * blockDim.x + threadIdx.x;
  if (i >= total) return;
  int d = (int)(i % DHD);
  int h = (int)((i / DHD) % HEADS);
  int j = (int)((i / DIMX) % JJ);
  int b = (int)(i / ((size_t)DIMX * JJ));
  Vt[(((size_t)(b * HEADS + h)) * DHD + d) * JJ + j] = vin[i];
}

// ---------------- generic WMMA GEMM (async double-buffered) ----------------
// C[M,Nc] = A_bf16[M,K] @ Bt_bf16[Nc,K]^T (+bias) (+Cin fp32)
template <int ACCUM, int OUT_BF16, int HAS_BIAS>
__global__ __launch_bounds__(256) void gemm_kernel(
    const unsigned short* __restrict__ A, const unsigned short* __restrict__ Bt,
    const float* __restrict__ bias, const float* __restrict__ Cin,
    void* __restrict__ Cout, int M, int Nc, int K) {
  __shared__ __align__(16) unsigned short As[2][128][80];
  __shared__ __align__(16) unsigned short Bs[2][128][80];
  const int m0 = blockIdx.y * 128, n0 = blockIdx.x * 128;
  const int t = threadIdx.x, lane = t & 31, wid = t >> 5;
  const int waveM = wid >> 2, waveN = wid & 3;
  const int lr = lane & 15, lh = lane >> 4;

  fvec8 c[4][2];
#pragma unroll
  for (int i = 0; i < 4; i++)
#pragma unroll
    for (int j = 0; j < 2; j++) c[i][j] = fzero();

  const int arow = t >> 1;
  const int acol = (t & 1) * 32;
  const unsigned short* agbase = A + (size_t)(m0 + arow) * K + acol;
  const unsigned short* bgbase = Bt + (size_t)(n0 + arow) * K + acol;

  // prologue: stage 0 into buffer 0 (8 async wave-instructions)
#pragma unroll
  for (int s = 0; s < 4; s++) {
    async_load_b128(agbase + s * 8, &As[0][arow][acol + s * 8]);
    async_load_b128(bgbase + s * 8, &Bs[0][arow][acol + s * 8]);
  }

  int buf = 0;
  for (int kb = 0; kb < K; kb += 64) {
    if (kb + 64 < K) {
      // issue next stage into the other buffer, then wait for current (<=8 left)
#pragma unroll
      for (int s = 0; s < 4; s++) {
        async_load_b128(agbase + kb + 64 + s * 8, &As[buf ^ 1][arow][acol + s * 8]);
        async_load_b128(bgbase + kb + 64 + s * 8, &Bs[buf ^ 1][arow][acol + s * 8]);
      }
      wait_asynccnt<8>();
    } else {
      wait_asynccnt<0>();
    }
    __syncthreads();
#pragma unroll
    for (int kk = 0; kk < 64; kk += 32) {
      FragB af[4], bf[2];
#pragma unroll
      for (int i = 0; i < 4; i++) {
        const unsigned short* p = &As[buf][waveM * 64 + 16 * i + lr][kk + lh * 8];
        af[i].q[0] = *(const uint4*)p;
        af[i].q[1] = *(const uint4*)(p + 16);
      }
#pragma unroll
      for (int j = 0; j < 2; j++) {
        const unsigned short* p = &Bs[buf][waveN * 32 + 16 * j + lr][kk + lh * 8];
        bf[j].q[0] = *(const uint4*)p;
        bf[j].q[1] = *(const uint4*)(p + 16);
      }
#pragma unroll
      for (int i = 0; i < 4; i++)
#pragma unroll
        for (int j = 0; j < 2; j++) c[i][j] = wmma_bf16(af[i], bf[j], c[i][j]);
    }
    __syncthreads();
    buf ^= 1;
  }

#pragma unroll
  for (int i = 0; i < 4; i++) {
#pragma unroll
    for (int j = 0; j < 2; j++) {
      int col = n0 + waveN * 32 + 16 * j + lr;
      float bv = HAS_BIAS ? bias[col] : 0.f;
#pragma unroll
      for (int r = 0; r < 8; r++) {
        int row = m0 + waveM * 64 + 16 * i + r + lh * 8;
        size_t idx = (size_t)row * Nc + col;
        float val = c[i][j][r] + bv;
        if (ACCUM) val += Cin[idx];
        if (OUT_BF16)
          ((unsigned short*)Cout)[idx] = f2bf(val);
        else
          ((float*)Cout)[idx] = val;
      }
    }
  }
}

// ---------------- fused GEGLU GEMM (async double-buffered) ----------------
// Y[M,4096] = (A@W1t[c] + b1[c]) * gelu(A@W1t[c+4096] + b1[c+4096])
__global__ __launch_bounds__(256) void geglu_kernel(
    const unsigned short* __restrict__ A, const unsigned short* __restrict__ W1t,
    const float* __restrict__ b1, unsigned short* __restrict__ Y, int K) {
  __shared__ __align__(16) unsigned short As[2][128][80];
  __shared__ __align__(16) unsigned short Ba[2][64][80];
  __shared__ __align__(16) unsigned short Bg[2][64][80];
  const int m0 = blockIdx.y * 128, c0 = blockIdx.x * 64;
  const int t = threadIdx.x, lane = t & 31, wid = t >> 5;
  const int waveM = wid >> 2, waveN = wid & 3;
  const int lr = lane & 15, lh = lane >> 4;

  fvec8 ca[4], cg[4];
#pragma unroll
  for (int i = 0; i < 4; i++) { ca[i] = fzero(); cg[i] = fzero(); }

  const int arow = t >> 1, acol = (t & 1) * 32;
  const int brow = t >> 2, bcol = (t & 3) * 16;
  const unsigned short* agbase  = A + (size_t)(m0 + arow) * K + acol;
  const unsigned short* babase  = W1t + (size_t)(c0 + brow) * K + bcol;
  const unsigned short* bgbase  = W1t + (size_t)(FFH + c0 + brow) * K + bcol;

  // prologue: stage 0 (4 + 2 + 2 = 8 async wave-instructions)
#pragma unroll
  for (int s = 0; s < 4; s++)
    async_load_b128(agbase + s * 8, &As[0][arow][acol + s * 8]);
#pragma unroll
  for (int s = 0; s < 2; s++) {
    async_load_b128(babase + s * 8, &Ba[0][brow][bcol + s * 8]);
    async_load_b128(bgbase + s * 8, &Bg[0][brow][bcol + s * 8]);
  }

  int buf = 0;
  for (int kb = 0; kb < K; kb += 64) {
    if (kb + 64 < K) {
#pragma unroll
      for (int s = 0; s < 4; s++)
        async_load_b128(agbase + kb + 64 + s * 8, &As[buf ^ 1][arow][acol + s * 8]);
#pragma unroll
      for (int s = 0; s < 2; s++) {
        async_load_b128(babase + kb + 64 + s * 8, &Ba[buf ^ 1][brow][bcol + s * 8]);
        async_load_b128(bgbase + kb + 64 + s * 8, &Bg[buf ^ 1][brow][bcol + s * 8]);
      }
      wait_asynccnt<8>();
    } else {
      wait_asynccnt<0>();
    }
    __syncthreads();
#pragma unroll
    for (int kk = 0; kk < 64; kk += 32) {
      FragB af[4], bfa, bfg;
#pragma unroll
      for (int i = 0; i < 4; i++) {
        const unsigned short* p = &As[buf][waveM * 64 + 16 * i + lr][kk + lh * 8];
        af[i].q[0] = *(const uint4*)p;
        af[i].q[1] = *(const uint4*)(p + 16);
      }
      {
        const unsigned short* p = &Ba[buf][waveN * 16 + lr][kk + lh * 8];
        bfa.q[0] = *(const uint4*)p;
        bfa.q[1] = *(const uint4*)(p + 16);
        const unsigned short* p2 = &Bg[buf][waveN * 16 + lr][kk + lh * 8];
        bfg.q[0] = *(const uint4*)p2;
        bfg.q[1] = *(const uint4*)(p2 + 16);
      }
#pragma unroll
      for (int i = 0; i < 4; i++) {
        ca[i] = wmma_bf16(af[i], bfa, ca[i]);
        cg[i] = wmma_bf16(af[i], bfg, cg[i]);
      }
    }
    __syncthreads();
    buf ^= 1;
  }

  int col = c0 + waveN * 16 + lr;
  float ba_ = b1[col];
  float bg_ = b1[FFH + col];
#pragma unroll
  for (int i = 0; i < 4; i++) {
#pragma unroll
    for (int r = 0; r < 8; r++) {
      int row = m0 + waveM * 64 + 16 * i + r + lh * 8;
      float a = ca[i][r] + ba_;
      float g = cg[i][r] + bg_;
      float gel = 0.5f * g * (1.f + erff(g * 0.70710678118f));
      Y[(size_t)row * FFH + col] = f2bf(a * gel);
    }
  }
}

// ---------------- flash attention with WMMA ----------------
// Q [B,H,N,DH], K [B,H,J,DH], Vt [B,H,DH,J] (bf16) -> O bf16 [B,N,H*DH]
__global__ __launch_bounds__(128) void attn_kernel(
    const unsigned short* __restrict__ Q, const unsigned short* __restrict__ Kk,
    const unsigned short* __restrict__ Vt, unsigned short* __restrict__ O) {
  __shared__ __align__(16) float Ssc[4][16][64];
  __shared__ __align__(16) unsigned short Pp[4][16][64];
  __shared__ float rmax[4][16], rsum[4][16], corr[4][16];

  const int blk = blockIdx.x;
  const int b = blk >> 6;          // / (HEADS*4)
  const int h = (blk >> 2) & 15;
  const int nt = blk & 3;
  const int t = threadIdx.x, lane = t & 31, wid = t >> 5;
  const int lr = lane & 15, lh = lane >> 4;

  // load Q A-fragments (reused for all j chunks)
  FragB qa[2];
  {
    size_t qbase = (((size_t)(b * HEADS + h)) * NN + nt * 64 + wid * 16 + lr) * DHD;
#pragma unroll
    for (int f = 0; f < 2; f++) {
      qa[f].q[0] = *(const uint4*)(Q + qbase + f * 32 + lh * 8);
      qa[f].q[1] = *(const uint4*)(Q + qbase + f * 32 + lh * 8 + 16);
    }
  }

  fvec8 of[4];
#pragma unroll
  for (int f = 0; f < 4; f++) of[f] = fzero();
  if (lane < 16) { rmax[wid][lr] = -3.0e38f; rsum[wid][lr] = 0.f; }
  __syncthreads();

  for (int j0 = 0; j0 < JJ; j0 += 64) {
    // scores for 16 rows x 64 cols
    fvec8 s[4];
#pragma unroll
    for (int jt = 0; jt < 4; jt++) {
      size_t kbase = (((size_t)(b * HEADS + h)) * JJ + j0 + jt * 16 + lr) * DHD;
      FragB kb0, kb1;
      kb0.q[0] = *(const uint4*)(Kk + kbase + lh * 8);
      kb0.q[1] = *(const uint4*)(Kk + kbase + lh * 8 + 16);
      kb1.q[0] = *(const uint4*)(Kk + kbase + 32 + lh * 8);
      kb1.q[1] = *(const uint4*)(Kk + kbase + 32 + lh * 8 + 16);
      fvec8 acc = fzero();
      acc = wmma_bf16(qa[0], kb0, acc);
      acc = wmma_bf16(qa[1], kb1, acc);
      s[jt] = acc;
    }
#pragma unroll
    for (int jt = 0; jt < 4; jt++)
#pragma unroll
      for (int r = 0; r < 8; r++)
        Ssc[wid][r + lh * 8][jt * 16 + lr] = s[jt][r] * SCALE;
    __syncthreads();

    // online softmax, one lane per row
    if (lane < 16) {
      int row = lr;
      float mo = rmax[wid][row];
      float mx = mo;
      for (int jj = 0; jj < 64; jj++) mx = fmaxf(mx, Ssc[wid][row][jj]);
      float cf = __expf(mo - mx);
      float sum = rsum[wid][row] * cf;
      for (int jj = 0; jj < 64; jj++) {
        float p = __expf(Ssc[wid][row][jj] - mx);
        sum += p;
        Pp[wid][row][jj] = f2bf(p);
      }
      rmax[wid][row] = mx;
      rsum[wid][row] = sum;
      corr[wid][row] = cf;
    }
    __syncthreads();

    // rescale O and accumulate P @ V
#pragma unroll
    for (int f = 0; f < 4; f++)
#pragma unroll
      for (int r = 0; r < 8; r++) of[f][r] *= corr[wid][r + lh * 8];

    FragB pa[2];
#pragma unroll
    for (int kk = 0; kk < 2; kk++) {
      const unsigned short* pp = &Pp[wid][lr][kk * 32 + lh * 8];
      pa[kk].q[0] = *(const uint4*)pp;
      pa[kk].q[1] = *(const uint4*)(pp + 16);
    }
#pragma unroll
    for (int f = 0; f < 4; f++) {
      size_t vb = (((size_t)(b * HEADS + h)) * DHD + f * 16 + lr) * JJ + j0;
      FragB v0, v1;
      v0.q[0] = *(const uint4*)(Vt + vb + lh * 8);
      v0.q[1] = *(const uint4*)(Vt + vb + lh * 8 + 16);
      v1.q[0] = *(const uint4*)(Vt + vb + 32 + lh * 8);
      v1.q[1] = *(const uint4*)(Vt + vb + 32 + lh * 8 + 16);
      of[f] = wmma_bf16(pa[0], v0, of[f]);
      of[f] = wmma_bf16(pa[1], v1, of[f]);
    }
    __syncthreads();
  }

#pragma unroll
  for (int f = 0; f < 4; f++) {
#pragma unroll
    for (int r = 0; r < 8; r++) {
      float inv = 1.f / rsum[wid][r + lh * 8];
      int n = nt * 64 + wid * 16 + r + lh * 8;
      int col = h * DHD + f * 16 + lr;
      O[((size_t)b * NN + n) * DIMX + col] = f2bf(of[f][r] * inv);
    }
  }
}

// ---------------- host orchestration ----------------
static inline int blocks_for(size_t n) { return (int)((n + 255) / 256); }

extern "C" void kernel_launch(void* const* d_in, const int* in_sizes, int n_in,
                              void* d_out, int out_size, void* d_ws, size_t ws_size,
                              hipStream_t stream) {
  (void)in_sizes; (void)n_in; (void)out_size; (void)ws_size;
  const float* x    = (const float*)d_in[0];
  const float* cond = (const float*)d_in[1];
  const float* Wq   = (const float*)d_in[2];
  const float* Wk   = (const float*)d_in[3];
  const float* Wv   = (const float*)d_in[4];
  const float* Wo   = (const float*)d_in[5];
  const float* bo   = (const float*)d_in[6];
  const float* rpe  = (const float*)d_in[7];
  const float* W1   = (const float*)d_in[8];
  const float* b1   = (const float*)d_in[9];
  const float* W2   = (const float*)d_in[10];
  const float* b2   = (const float*)d_in[11];
  const float* src_freq = (const float*)d_in[12];
  const float* tgt_freq = (const float*)d_in[13];
  const int*   rel_idx  = (const int*)d_in[14];

  char* ws = (char*)d_ws;
  size_t off = 0;
  auto carve = [&](size_t bytes) {
    char* p = ws + off;
    off += (bytes + 255) & ~(size_t)255;
    return p;
  };
  const size_t MQ = (size_t)BB * NN;   // 2048 rows
  const size_t MK = (size_t)BB * JJ;   // 8704 rows

  float*          xbuf   = (float*)carve(MQ * DIMX * 4);
  unsigned short* xbf    = (unsigned short*)carve(MQ * DIMX * 2);
  unsigned short* condbf = (unsigned short*)carve(MK * DIMX * 2);
  unsigned short* wbuf   = (unsigned short*)carve((size_t)8192 * 1024 * 2);
  float*          qf32   = (float*)carve(MQ * DIMX * 4);
  float*          kf32   = (float*)carve(MK * DIMX * 4);
  unsigned short* vbf    = (unsigned short*)carve(MK * DIMX * 2);
  unsigned short* qbf    = (unsigned short*)carve(MQ * DIMX * 2);
  unsigned short* kbf    = (unsigned short*)carve(MK * DIMX * 2);
  unsigned short* vT     = (unsigned short*)carve(MK * DIMX * 2);
  unsigned short* aobf   = (unsigned short*)carve(MQ * DIMX * 2);
  unsigned short* ybf    = (unsigned short*)carve(MQ * FFH * 2);

  // residual stream init + one-time cond conversion
  (void)hipMemcpyAsync(xbuf, x, MQ * DIMX * 4, hipMemcpyDeviceToDevice, stream);
  f2bf_copy_kernel<<<blocks_for(MK * DIMX), 256, 0, stream>>>(cond, condbf, MK * DIMX);

  for (int l = 0; l < DEPTH; l++) {
    const float* Wq_l = Wq + (size_t)l * DIMX * DIMX;
    const float* Wk_l = Wk + (size_t)l * DIMX * DIMX;
    const float* Wv_l = Wv + (size_t)l * DIMX * DIMX;
    const float* Wo_l = Wo + (size_t)l * DIMX * DIMX;
    const float* bo_l = bo + (size_t)l * DIMX;
    const float* rpe_l = rpe + (size_t)l * 405 * HEADS;
    const float* W1_l = W1 + (size_t)l * DIMX * (2 * FFH);
    const float* b1_l = b1 + (size_t)l * (2 * FFH);
    const float* W2_l = W2 + (size_t)l * FFH * DIMX;
    const float* b2_l = b2 + (size_t)l * DIMX;

    f2bf_copy_kernel<<<blocks_for(MQ * DIMX), 256, 0, stream>>>(xbuf, xbf, MQ * DIMX);

    // Q = x @ Wq
    f2bf_transpose_kernel<<<blocks_for((size_t)DIMX * DIMX), 256, 0, stream>>>(Wq_l, wbuf, DIMX, DIMX);
    gemm_kernel<0, 0, 0><<<dim3(DIMX / 128, (int)(MQ / 128)), 256, 0, stream>>>(
        xbf, wbuf, nullptr, nullptr, qf32, (int)MQ, DIMX, DIMX);
    // K = cond @ Wk
    f2bf_transpose_kernel<<<blocks_for((size_t)DIMX * DIMX), 256, 0, stream>>>(Wk_l, wbuf, DIMX, DIMX);
    gemm_kernel<0, 0, 0><<<dim3(DIMX / 128, (int)(MK / 128)), 256, 0, stream>>>(
        condbf, wbuf, nullptr, nullptr, kf32, (int)MK, DIMX, DIMX);
    // V = cond @ Wv (bf16 out)
    f2bf_transpose_kernel<<<blocks_for((size_t)DIMX * DIMX), 256, 0, stream>>>(Wv_l, wbuf, DIMX, DIMX);
    gemm_kernel<0, 1, 0><<<dim3(DIMX / 128, (int)(MK / 128)), 256, 0, stream>>>(
        condbf, wbuf, nullptr, nullptr, vbf, (int)MK, DIMX, DIMX);

    vtrans_kernel<<<blocks_for(MK * DIMX), 256, 0, stream>>>(vbf, vT);
    rope_q_kernel<<<blocks_for((size_t)BB * NN * HEADS * (DHD / 2)), 256, 0, stream>>>(
        qf32, src_freq, qbf);
    biasrope_k_kernel<<<blocks_for((size_t)BB * JJ * HEADS * (DHD / 2)), 256, 0, stream>>>(
        kf32, tgt_freq, rpe_l, rel_idx, kbf);

    attn_kernel<<<BB * HEADS * 4, 128, 0, stream>>>(qbf, kbf, vT, aobf);

    // x += attn_out @ Wo + bo
    f2bf_transpose_kernel<<<blocks_for((size_t)DIMX * DIMX), 256, 0, stream>>>(Wo_l, wbuf, DIMX, DIMX);
    gemm_kernel<1, 0, 1><<<dim3(DIMX / 128, (int)(MQ / 128)), 256, 0, stream>>>(
        aobf, wbuf, bo_l, xbuf, xbuf, (int)MQ, DIMX, DIMX);

    // GEGLU FFN
    f2bf_copy_kernel<<<blocks_for(MQ * DIMX), 256, 0, stream>>>(xbuf, xbf, MQ * DIMX);
    f2bf_transpose_kernel<<<blocks_for((size_t)DIMX * 2 * FFH), 256, 0, stream>>>(W1_l, wbuf, DIMX, 2 * FFH);
    geglu_kernel<<<dim3(FFH / 64, (int)(MQ / 128)), 256, 0, stream>>>(xbf, wbuf, b1_l, ybf, DIMX);
    f2bf_transpose_kernel<<<blocks_for((size_t)FFH * DIMX), 256, 0, stream>>>(W2_l, wbuf, FFH, DIMX);
    gemm_kernel<1, 0, 1><<<dim3(DIMX / 128, (int)(MQ / 128)), 256, 0, stream>>>(
        ybf, wbuf, b2_l, xbuf, xbuf, (int)MQ, DIMX, FFH);
  }

  (void)hipMemcpyAsync(d_out, xbuf, MQ * DIMX * 4, hipMemcpyDeviceToDevice, stream);
}
